// FRCLoss_38826504355868
// MI455X (gfx1250) — compile-verified, matrix operand order checked by reading, not run on hardware
//
#include <hip/hip_runtime.h>
#include <math.h>

// ---------------------------------------------------------------------------
// FRC loss on MI455X (gfx1250, wave32).
// 2-D real FFT done as f16 WMMA matmuls with f32 accumulation:
//   H   = A * F        (row DFT, keep k = 0..128, padded to 144)
//   FA  = F * H        (col DFT; F is symmetric so all fetches are row-major)
// then per-ring sums via LDS ds_add_f32 atomics, global flush, scalar FRC.
// ---------------------------------------------------------------------------

typedef __attribute__((ext_vector_type(16))) _Float16 v16h;
typedef __attribute__((ext_vector_type(8)))  _Float16 v8h;
typedef __attribute__((ext_vector_type(8)))  float    v8f;

union V16 { v16h v; v8h h[2]; };

__device__ __forceinline__ v16h ld16(const _Float16* p0, const _Float16* p1) {
    V16 u;
    u.h[0] = *(const v8h*)p0;
    u.h[1] = *(const v8h*)p1;
    return u.v;
}

#define WMMA_F16(A, B, C) \
    __builtin_amdgcn_wmma_f32_16x16x32_f16(false, (A), false, (B), (short)0, (C), false, false)

// Workspace layout (bytes):
//   ftab : f16 [2][256][256]  twiddles (cos, sin of -2*pi*n*k/256)   @ 0        (256 KiB)
//   sa   : f32 [256][128]     per-a-image ring |FA|^2                @ 262144
//   sb   : f32 [256][128]     per-b-image ring |FB|^2                @ 393216
//   sab  : f32 [256][128]     per-pair ring Re(FA conj FB)           @ 524288
//   H    : f16 [64 slots][2 comps][144 k][256 r] (transposed)        @ 655360   (9.0 MiB)
#define FTAB_OFF  0
#define SA_OFF    262144
#define SB_OFF    393216
#define SAB_OFF   524288
#define H_OFF     655360
#define H_SLOT_HALFS (2 * 144 * 256)

// ---------------------------------------------------------------------------
// Init: build twiddle table with exact mod-256 angle reduction; zero ring bins.
// ---------------------------------------------------------------------------
__global__ __launch_bounds__(256) void frc_init(_Float16* __restrict__ ftab,
                                                float* __restrict__ bins3) {
    int t = blockIdx.x * blockDim.x + threadIdx.x;
    if (t < 65536) {
        int n = t >> 8, k = t & 255;
        int ph = (n * k) & 255;                       // exact periodic reduction
        float ang = -6.283185307179586f * (float)ph * (1.0f / 256.0f);
        ftab[t]          = (_Float16)__cosf(ang);     // Fr
        ftab[65536 + t]  = (_Float16)__sinf(ang);     // Fi
    }
    if (t < 98304) bins3[t] = 0.0f;                   // sa|sb|sab contiguous
}

// ---------------------------------------------------------------------------
// Stage 1: row DFT.  grid = (64 image-slots, 16 row-blocks), 288 thr = 9 waves.
// Wave w handles k-tile w (k = 16w .. 16w+15).  H stored transposed: H_T[k][r].
// ---------------------------------------------------------------------------
__global__ __launch_bounds__(288) void frc_stage1(const float* __restrict__ a_images,
                                                  const float* __restrict__ b_images,
                                                  const _Float16* __restrict__ ftab,
                                                  _Float16* __restrict__ Hbuf,
                                                  int chunk) {
    __shared__ _Float16 ldsA[16 * 256];               // 8 KiB f16 image strip

    const int slot = blockIdx.x;                      // 0..63
    const int pair = chunk * 32 + (slot >> 1);
    const int set  = slot & 1;
    const int rblk = blockIdx.y;                      // 0..15
    const float* img = (set == 0 ? a_images : b_images)
                       + (size_t)pair * 65536 + (size_t)rblk * 16 * 256;

    // Stage strip to LDS once (f32 -> f16), shared by all 9 waves.
    for (int t = threadIdx.x; t < 16 * 256; t += 288)
        ldsA[t] = (_Float16)img[t];
    __syncthreads();

    const int wave = threadIdx.x >> 5;                // k-tile 0..8
    const int lane = threadIdx.x & 31;
    const int nl   = lane & 15;
    const int hh   = lane >> 4;
    const int kcol = wave * 16 + nl;

    const _Float16* fr = ftab + (size_t)kcol * 256;          // symmetric: row kcol
    const _Float16* fi = fr + 65536;

    v8f accR = {}; v8f accI = {};
    #pragma unroll
    for (int c = 0; c < 8; ++c) {                     // K = 256 in chunks of 32
        const int ko = c * 32 + hh * 8;
        v16h av = ld16(&ldsA[nl * 256 + ko], &ldsA[nl * 256 + ko + 16]);
        v16h br = ld16(fr + ko, fr + ko + 16);
        v16h bi = ld16(fi + ko, fi + ko + 16);
        if (c < 7) __builtin_prefetch(fr + ko + 32, 0, 3);
        accR = WMMA_F16(av, br, accR);
        accI = WMMA_F16(av, bi, accI);
    }

    // D tile: M = j + 8*hh (row r), N = nl (freq k).  Store transposed H_T[k][r].
    _Float16* Hr = Hbuf + (size_t)slot * H_SLOT_HALFS + (size_t)kcol * 256 + rblk * 16;
    _Float16* Hi = Hr + 144 * 256;
    #pragma unroll
    for (int j = 0; j < 8; ++j) {
        Hr[j + 8 * hh] = (_Float16)accR[j];
        Hi[j + 8 * hh] = (_Float16)accI[j];
    }
}

// ---------------------------------------------------------------------------
// Stage 2: column DFT + ring accumulation.
// grid = (32 pairs, 16 m-blocks), 288 thr = 9 waves (one k-tile each).
//   FAr = Fr*Har - Fi*Hai ; FAi = Fr*Hai + Fi*Har   (same for image b)
// ---------------------------------------------------------------------------
__global__ __launch_bounds__(288) void frc_stage2(const _Float16* __restrict__ ftab,
                                                  const _Float16* __restrict__ Hbuf,
                                                  const int* __restrict__ rr_ids,
                                                  float* __restrict__ sa,
                                                  float* __restrict__ sb,
                                                  float* __restrict__ sab,
                                                  int chunk) {
    __shared__ float bins[3 * 128];

    const int pslot = blockIdx.x;                     // 0..31
    const int pair  = chunk * 32 + pslot;
    const int mblk  = blockIdx.y;                     // 0..15
    for (int t = threadIdx.x; t < 384; t += 288) bins[t] = 0.0f;
    __syncthreads();

    const int lane = threadIdx.x & 31;
    const int nl   = lane & 15;
    const int hh   = lane >> 4;
    const int kcol = (threadIdx.x >> 5) * 16 + nl;    // wave picks k-tile

    const _Float16* HaR = Hbuf + (size_t)(2 * pslot) * H_SLOT_HALFS + (size_t)kcol * 256;
    const _Float16* HaI = HaR + 144 * 256;
    const _Float16* HbR = HaR + H_SLOT_HALFS;
    const _Float16* HbI = HbR + 144 * 256;
    const _Float16* frRow = ftab + (size_t)(mblk * 16 + nl) * 256;   // A: Fsym[m][r]
    const _Float16* fiRow = frRow + 65536;

    v8f ar = {}, ai = {}, br_ = {}, bi_ = {};
    #pragma unroll
    for (int c = 0; c < 8; ++c) {                     // K = r = 256 in chunks of 32
        const int ko = c * 32 + hh * 8;
        v16h Fr  = ld16(frRow + ko, frRow + ko + 16);
        v16h Fi  = ld16(fiRow + ko, fiRow + ko + 16);
        v16h Fin = -Fi;
        v16h har = ld16(HaR + ko, HaR + ko + 16);
        v16h hai = ld16(HaI + ko, HaI + ko + 16);
        v16h hbr = ld16(HbR + ko, HbR + ko + 16);
        v16h hbi = ld16(HbI + ko, HbI + ko + 16);
        ar  = WMMA_F16(Fr,  har, ar);  ar  = WMMA_F16(Fin, hai, ar);
        ai  = WMMA_F16(Fr,  hai, ai);  ai  = WMMA_F16(Fi,  har, ai);
        br_ = WMMA_F16(Fr,  hbr, br_); br_ = WMMA_F16(Fin, hbi, br_);
        bi_ = WMMA_F16(Fr,  hbi, bi_); bi_ = WMMA_F16(Fi,  hbr, bi_);
    }

    // Ring accumulation (divergence only here, after all WMMAs: EXEC all-1 above).
    if (kcol < 129) {
        const int* rr = rr_ids + (size_t)(mblk * 16) * 129 + kcol;
        #pragma unroll
        for (int j = 0; j < 8; ++j) {
            const int m   = j + 8 * hh;               // D tile: M = j + 8*hh
            const int bin = rr[(size_t)m * 129];
            if (bin < 128) {
                const float far = ar[j],  fai = ai[j];
                const float fbr = br_[j], fbi = bi_[j];
                atomicAdd(&bins[bin],       far * far + fai * fai);
                atomicAdd(&bins[128 + bin], fbr * fbr + fbi * fbi);
                atomicAdd(&bins[256 + bin], far * fbr + fai * fbi);
            }
        }
    }
    __syncthreads();

    for (int t = threadIdx.x; t < 384; t += 288) {
        float v = bins[t];
        float* dst = (t < 128) ? &sa[(size_t)pair * 128 + t]
                   : (t < 256) ? &sb[(size_t)pair * 128 + (t - 128)]
                               : &sab[(size_t)pair * 128 + (t - 256)];
        atomicAdd(dst, v);
    }
}

// ---------------------------------------------------------------------------
// Final: frc = sab / max(sqrt(sa)*sqrt(sb), 1e-4), mean over bins 1..126, pairs.
// ---------------------------------------------------------------------------
__global__ __launch_bounds__(256) void frc_final(const float* __restrict__ sa,
                                                 const float* __restrict__ sb,
                                                 const float* __restrict__ sab,
                                                 float* __restrict__ out) {
    __shared__ float red[256];
    const int p = threadIdx.x;                        // pair index
    float s = 0.0f;
    for (int bin = 1; bin <= 126; ++bin) {
        const float den = fmaxf(sqrtf(sa[p * 128 + bin]) * sqrtf(sb[p * 128 + bin]), 1e-4f);
        s += sab[p * 128 + bin] / den;
    }
    red[p] = s;
    __syncthreads();
    for (int off = 128; off > 0; off >>= 1) {
        if (p < off) red[p] += red[p + off];
        __syncthreads();
    }
    if (p == 0) out[0] = red[0] / (256.0f * 126.0f);
}

// ---------------------------------------------------------------------------
extern "C" void kernel_launch(void* const* d_in, const int* in_sizes, int n_in,
                              void* d_out, int out_size, void* d_ws, size_t ws_size,
                              hipStream_t stream) {
    (void)in_sizes; (void)n_in; (void)out_size; (void)ws_size;
    const float* a  = (const float*)d_in[0];
    const float* b  = (const float*)d_in[1];
    const int*   rr = (const int*)d_in[2];

    char* ws = (char*)d_ws;
    _Float16* ftab = (_Float16*)(ws + FTAB_OFF);
    float*    sa   = (float*)(ws + SA_OFF);
    float*    sb   = (float*)(ws + SB_OFF);
    float*    sab  = (float*)(ws + SAB_OFF);
    _Float16* Hbuf = (_Float16*)(ws + H_OFF);

    // Build twiddles + zero ring bins (sa/sb/sab are contiguous from SA_OFF).
    frc_init<<<384, 256, 0, stream>>>(ftab, sa);

    // 8 chunks of 32 pairs: H slice (9.4 MB) stays L2-resident between stages.
    for (int chunk = 0; chunk < 8; ++chunk) {
        frc_stage1<<<dim3(64, 16), 288, 0, stream>>>(a, b, ftab, Hbuf, chunk);
        frc_stage2<<<dim3(32, 16), 288, 0, stream>>>(ftab, Hbuf, rr, sa, sb, sab, chunk);
    }

    frc_final<<<1, 256, 0, stream>>>(sa, sb, sab, (float*)d_out);
}
